// SparseAttention_72112500900081
// MI455X (gfx1250) — matrix operand order repaired and checked by reading, not compiled, not run on hardware
//
#include <hip/hip_runtime.h>
#include <stdint.h>

// ---------------------------------------------------------------------------
// MI455X (gfx1250, CDNA5) sparse attention.
// Precision: fp32 inputs converted once to bf16; all GEMMs via
// V_WMMA_F32_16X16X32_BF16 (wave32), f32 accumulate.
// Both big GEMMs use CDNA5 async global->LDS loads (ASYNCcnt) with
// double-buffered K pipelines (unroll-2 so buffer selects are static).
// ---------------------------------------------------------------------------

typedef __attribute__((ext_vector_type(16))) __bf16 v16bf;
typedef __attribute__((ext_vector_type(8)))  float  v8f;

#define B_   8
#define S_   4096
#define D_   1024
#define H_   16
#define DH_  64
#define KSEL 1024

static __device__ __forceinline__ uint16_t f2bf(float f) {
  uint32_t u = __float_as_uint(f);
  uint32_t r = 0x7FFFu + ((u >> 16) & 1u);     // round-to-nearest-even
  return (uint16_t)((u + r) >> 16);
}

union FragU { v16bf v; uint4 u[2]; };

// Two contiguous 16-byte chunks -> one 16-element bf16 fragment (8 VGPRs).
static __device__ __forceinline__ v16bf load_frag(const uint16_t* p0,
                                                  const uint16_t* p1) {
  FragU f;
  f.u[0] = *(const uint4*)p0;
  f.u[1] = *(const uint4*)p1;
  return f.v;
}

static __device__ __forceinline__ v8f wmma_bf16(v16bf a, v16bf b, v8f c) {
  // (neg_a, A, neg_b, B, c_mod, C, reuse_a, reuse_b)
  return __builtin_amdgcn_wmma_f32_16x16x32_bf16(false, a, false, b,
                                                 (short)0, c, false, false);
}

static __device__ __forceinline__ v8f zero_v8f() {
  v8f z;
  #pragma unroll
  for (int e = 0; e < 8; ++e) z[e] = 0.f;
  return z;
}

// --- CDNA5 async global->LDS copy: 32 bytes per lane (two B128 issues).
// LDS destination is the low 32 bits of the generic shared-aperture address
//   Tracked with ASYNCcnt.
static __device__ __forceinline__ void async_copy_32B(uint32_t lds_addr,
                                                      const void* gptr) {
  uint64_t ga = (uint64_t)(uintptr_t)gptr;
  asm volatile("global_load_async_to_lds_b128 %0, %1, off\n\t"
               "global_load_async_to_lds_b128 %0, %1, off offset:16"
               :: "v"(lds_addr), "v"(ga) : "memory");
}
static __device__ __forceinline__ void wait_async_le4() {
  asm volatile("s_wait_asynccnt 4" ::: "memory");
}
static __device__ __forceinline__ void wait_async_le2() {
  asm volatile("s_wait_asynccnt 2" ::: "memory");
}
static __device__ __forceinline__ void wait_async_0() {
  asm volatile("s_wait_asynccnt 0" ::: "memory");
}

// ---------------------------------------------------------------------------
// K1: squared row norms of query (ordering-equivalent to L2 norm for top-k).
// One wave per row, 8 waves per block.
// ---------------------------------------------------------------------------
__global__ void __launch_bounds__(256) norms_kernel(const float* __restrict__ q,
                                                    float* __restrict__ norms) {
  const int wave = threadIdx.x >> 5;
  const int lane = threadIdx.x & 31;
  const int row  = blockIdx.x * 8 + wave;
  const float4* p = (const float4*)(q + (size_t)row * D_);
  float acc = 0.f;
  #pragma unroll
  for (int i = lane; i < D_ / 4; i += 32) {
    float4 x = p[i];
    acc += x.x * x.x + x.y * x.y + x.z * x.z + x.w * x.w;
  }
  #pragma unroll
  for (int m = 16; m >= 1; m >>= 1) acc += __shfl_xor(acc, m, 32);
  if (lane == 0) norms[row] = acc;
}

// ---------------------------------------------------------------------------
// K2: exact top-1024 per batch: full bitonic sort of 4096 (val desc,
// index-ascending tiebreak to match jax.lax.top_k) in LDS.
// ---------------------------------------------------------------------------
__global__ void __launch_bounds__(1024) topk_kernel(const float* __restrict__ norms,
                                                    int* __restrict__ idx) {
  __shared__ float sv[S_];
  __shared__ int   si[S_];
  const int b = blockIdx.x;
  const int t = threadIdx.x;
  for (int i = t; i < S_; i += 1024) { sv[i] = norms[b * S_ + i]; si[i] = i; }
  __syncthreads();
  for (int k2 = 2; k2 <= S_; k2 <<= 1) {
    for (int j = k2 >> 1; j > 0; j >>= 1) {
      for (int i = t; i < S_; i += 1024) {
        const int ix = i ^ j;
        if (ix > i) {
          const bool descBlock = ((i & k2) == 0);
          float va = sv[i], vb = sv[ix];
          int   ia = si[i], ib = si[ix];
          const bool aFirst = (va > vb) || (va == vb && ia < ib); // "greater"
          const bool doSwap = descBlock ? (!aFirst) : aFirst;
          if (doSwap) { sv[i] = vb; sv[ix] = va; si[i] = ib; si[ix] = ia; }
        }
      }
      __syncthreads();
    }
  }
  if (t < KSEL) idx[b * KSEL + t] = si[t];
}

// ---------------------------------------------------------------------------
// K3: fp32 -> bf16 weight conversion.
// ---------------------------------------------------------------------------
__global__ void convert_bf16_kernel(const float* __restrict__ src,
                                    uint16_t* __restrict__ dst, int n) {
  int i = blockIdx.x * blockDim.x + threadIdx.x;
  if (i < n) dst[i] = f2bf(src[i]);
}

// ---------------------------------------------------------------------------
// K4: fused gather + input projections (double-buffered pipeline).
//   A rows are gathered from query/key/value (per N-tile sel), fp32->bf16
//   converted through VGPRs; next-stage A is register-prefetched during
//   current compute.  B = in_proj_w rows stream ASYNChronously into the
//   alternate LDS buffer (2 asyncs/thread/stage; s_wait_asynccnt 2 retires
//   exactly the current stage).
//   Outputs: qp/kp as (b,h,token,dh) bf16; v stored transposed vt
//   (b,h,dh,token) so the attention PV GEMM's B-operand is K-contiguous.
// Tile 128x128x32, 256 threads = 8 waves, each wave 64x32 (4x2 frags).
// ---------------------------------------------------------------------------
__global__ void __launch_bounds__(256) proj_gemm_kernel(
    const float* __restrict__ xq, const float* __restrict__ xk,
    const float* __restrict__ xv, const int* __restrict__ idx,
    const uint16_t* __restrict__ wIn,   // [3072][1024] bf16
    const float* __restrict__ bIn,      // [3072]
    uint16_t* __restrict__ qp, uint16_t* __restrict__ kp,
    uint16_t* __restrict__ vt) {
  __shared__ uint16_t aL[2][128][40];   // +8 pad: conflict avoidance
  __shared__ uint16_t bL[2][128][40];
  const uint32_t BUFB = 128 * 40 * 2;   // bytes per buffer

  const int tileM = blockIdx.x;       // 0..7   (tokens)
  const int tileN = blockIdx.y;       // 0..23  (3*1024 features)
  const int b     = blockIdx.z;       // 0..7
  const int tid  = threadIdx.x;
  const int lane = tid & 31, wave = tid >> 5;
  const int m16  = lane & 15, half = lane >> 4;
  const int mw   = wave >> 2;         // 0..1
  const int nw   = wave & 3;          // 0..3

  const int nBase = tileN * 128;
  const int sel   = nBase >> 10;      // 0:q 1:k 2:v (128 | 1024, never straddles)
  const float* __restrict__ x = (sel == 0) ? xq : (sel == 1) ? xk : xv;

  const int rowA = tid >> 1;              // 2 threads per tile row
  const int colA = (tid & 1) * 16;        // 16 elements each
  const int tokenA = idx[b * KSEL + tileM * 128 + rowA];
  const float* __restrict__ xrow = x + ((size_t)b * S_ + (size_t)tokenA) * D_;
  const uint16_t* __restrict__ wrow = wIn + (size_t)(nBase + rowA) * D_;
  const uint32_t bDst = (uint32_t)(uintptr_t)&bL[0][rowA][colA];

  v8f acc[4][2];
  #pragma unroll
  for (int i = 0; i < 4; ++i)
    #pragma unroll
    for (int j = 0; j < 2; ++j) acc[i][j] = zero_v8f();

  // prologue: stage-0 B async; stage-0 A into registers
  async_copy_32B(bDst, wrow + colA);
  float4 f0, f1, f2, f3;
  {
    const float4* fp = (const float4*)(xrow + colA);
    f0 = fp[0]; f1 = fp[1]; f2 = fp[2]; f3 = fp[3];
  }

  #pragma unroll 2
  for (int ks = 0; ks < 32; ++ks) {
    const int cur = ks & 1;
    // convert & store A(cur) to LDS
    uint32_t* ap = (uint32_t*)&aL[cur][rowA][colA];
    ap[0] = (uint32_t)f2bf(f0.x) | ((uint32_t)f2bf(f0.y) << 16);
    ap[1] = (uint32_t)f2bf(f0.z) | ((uint32_t)f2bf(f0.w) << 16);
    ap[2] = (uint32_t)f2bf(f1.x) | ((uint32_t)f2bf(f1.y) << 16);
    ap[3] = (uint32_t)f2bf(f1.z) | ((uint32_t)f2bf(f1.w) << 16);
    ap[4] = (uint32_t)f2bf(f2.x) | ((uint32_t)f2bf(f2.y) << 16);
    ap[5] = (uint32_t)f2bf(f2.z) | ((uint32_t)f2bf(f2.w) << 16);
    ap[6] = (uint32_t)f2bf(f3.x) | ((uint32_t)f2bf(f3.y) << 16);
    ap[7] = (uint32_t)f2bf(f3.z) | ((uint32_t)f2bf(f3.w) << 16);
    // kick off next stage
    if (ks < 31) {
      const int k0n = (ks + 1) * 32;
      const float4* fp = (const float4*)(xrow + k0n + colA);
      f0 = fp[0]; f1 = fp[1]; f2 = fp[2]; f3 = fp[3];   // A(next) -> regs
      async_copy_32B(bDst + (uint32_t)(cur ^ 1) * BUFB, wrow + k0n + colA);
      wait_async_le2();     // B(cur) retired (async loads complete in order)
    } else {
      wait_async_0();
    }
    __syncthreads();        // also drains DScnt for the A stores

    #pragma unroll
    for (int nf = 0; nf < 2; ++nf) {
      const uint16_t* bp = &bL[cur][nw * 32 + nf * 16 + m16][8 * half];
      v16bf bf = load_frag(bp, bp + 16);
      #pragma unroll
      for (int mf = 0; mf < 4; ++mf) {
        const uint16_t* apf = &aL[cur][mw * 64 + mf * 16 + m16][8 * half];
        v16bf af = load_frag(apf, apf + 16);
        acc[mf][nf] = wmma_bf16(af, bf, acc[mf][nf]);
      }
    }
    __syncthreads();        // all waves done reading `cur` before refill
  }

  #pragma unroll
  for (int mf = 0; mf < 4; ++mf)
    #pragma unroll
    for (int nf = 0; nf < 2; ++nf) {
      const int nLoc  = nw * 32 + nf * 16 + m16;
      const int nGlob = nBase + nLoc;
      const float bias = bIn[nGlob];
      const int nFeat = nGlob & 1023;
      const int h = nFeat >> 6, dh = nFeat & 63;
      #pragma unroll
      for (int r = 0; r < 8; ++r) {
        const int m = tileM * 128 + mw * 64 + mf * 16 + 8 * half + r;
        const uint16_t val = f2bf(acc[mf][nf][r] + bias);
        if (sel == 2) {
          vt[((size_t)(b * H_ + h) * DH_ + dh) * KSEL + m] = val;
        } else {
          const size_t off = ((size_t)(b * H_ + h) * KSEL + m) * DH_ + dh;
          if (sel == 0) qp[off] = val; else kp[off] = val;
        }
      }
    }
}

// ---------------------------------------------------------------------------
// K5: attention. Block owns (b, 16 q-rows), loops all 16 heads serially:
//   scores(16x1024) = q @ k^T * 1/8     (WMMA, f32 scores in LDS)
//   softmax rows (16-lane shfl reductions), probs -> bf16 in LDS,
//   attn_weights head-mean accumulated race-free (block owns its rows),
//   ctx(16x64) = P @ V via WMMA with vt (dh-major) as B-operand,
//   K split across wave pairs, reduced through LDS.
// Dynamic LDS ~101 KB (legal: up to 320 KB/WGP on CDNA5).
// ---------------------------------------------------------------------------
#define SSTR 1032   // f32 score row stride
#define PSTR 1040   // bf16 prob row stride

__global__ void __launch_bounds__(256) attn_kernel(
    const uint16_t* __restrict__ qp, const uint16_t* __restrict__ kp,
    const uint16_t* __restrict__ vt, uint16_t* __restrict__ ctx,
    float* __restrict__ attnw) {
  extern __shared__ char smem[];
  float*    sS   = (float*)smem;                               // [16][SSTR]
  uint16_t* sP   = (uint16_t*)(smem + 16 * SSTR * 4);          // [16][PSTR]
  float*    sRed = (float*)(smem + 16 * SSTR * 4 + 16 * PSTR * 2); // [4][256]

  const int qt = blockIdx.x;        // 0..63
  const int b  = blockIdx.y;        // 0..7
  const int qBase = qt * 16;
  const int tid = threadIdx.x;
  const int lane = tid & 31, wave = tid >> 5;
  const int m16 = lane & 15, half = lane >> 4;
  const float scale = 0.125f;       // 1/sqrt(64)

  for (int h = 0; h < H_; ++h) {
    const uint16_t* qh  = qp + (size_t)(b * H_ + h) * KSEL * DH_;
    const uint16_t* kh  = kp + (size_t)(b * H_ + h) * KSEL * DH_;
    const uint16_t* vth = vt + (size_t)(b * H_ + h) * DH_ * KSEL;

    // ---- scores: each wave covers 128 keys ----
    const uint16_t* aq = qh + (size_t)(qBase + m16) * DH_;
    v16bf a0 = load_frag(aq +      8 * half, aq + 16 + 8 * half); // dh 0..31
    v16bf a1 = load_frag(aq + 32 + 8 * half, aq + 48 + 8 * half); // dh 32..63
    #pragma unroll
    for (int nf = 0; nf < 8; ++nf) {
      const int n0 = wave * 128 + nf * 16;
      const uint16_t* bk = kh + (size_t)(n0 + m16) * DH_;
      v16bf b0 = load_frag(bk +      8 * half, bk + 16 + 8 * half);
      v16bf b1 = load_frag(bk + 32 + 8 * half, bk + 48 + 8 * half);
      v8f acc = zero_v8f();
      acc = wmma_bf16(a0, b0, acc);
      acc = wmma_bf16(a1, b1, acc);
      #pragma unroll
      for (int r = 0; r < 8; ++r)
        sS[(r + 8 * half) * SSTR + n0 + m16] = acc[r] * scale;
    }
    __syncthreads();

    // ---- softmax: 16 threads per row (shfl_xor width-16 groups) ----
    {
      const int row = tid >> 4;
      const int cg  = tid & 15;
      float* srow = sS + row * SSTR;
      float mx = -3.0e38f;
      for (int c = cg; c < KSEL; c += 16) mx = fmaxf(mx, srow[c]);
      #pragma unroll
      for (int m2 = 1; m2 < 16; m2 <<= 1) mx = fmaxf(mx, __shfl_xor(mx, m2, 16));
      float sum = 0.f;
      for (int c = cg; c < KSEL; c += 16) sum += __expf(srow[c] - mx);
      #pragma unroll
      for (int m2 = 1; m2 < 16; m2 <<= 1) sum += __shfl_xor(sum, m2, 16);
      const float inv = 1.0f / sum;
      float* awrow = attnw + ((size_t)b * KSEL + qBase + row) * KSEL;
      uint16_t* prow = sP + row * PSTR;
      for (int c = cg; c < KSEL; c += 16) {
        const float p = __expf(srow[c] - mx) * inv;
        prow[c] = f2bf(p);
        const float contrib = p * (1.0f / H_);
        if (h == 0) awrow[c] = contrib;       // block owns these rows:
        else        awrow[c] += contrib;      // race-free RMW across heads
      }
    }
    __syncthreads();

    // ---- ctx = P @ V : waves (nf = w&3, khalf = w>>2), reduce pairs ----
    {
      const int nf = wave & 3;
      const int kh2 = wave >> 2;
      v8f acc = zero_v8f();
      const uint16_t* btv = vth + (size_t)(nf * 16 + m16) * KSEL; // row = dh
      #pragma unroll 4
      for (int ks = 0; ks < 16; ++ks) {
        const int k0 = kh2 * 512 + ks * 32;
        const uint16_t* ap = sP + m16 * PSTR + k0;
        v16bf af = load_frag(ap + 8 * half, ap + 16 + 8 * half);
        v16bf bf = load_frag(btv + k0 + 8 * half, btv + k0 + 16 + 8 * half);
        acc = wmma_bf16(af, bf, acc);
      }
      if (kh2 == 1) {
        #pragma unroll
        for (int r = 0; r < 8; ++r) sRed[nf * 256 + r * 32 + lane] = acc[r];
      }
      __syncthreads();
      if (kh2 == 0) {
        #pragma unroll
        for (int r = 0; r < 8; ++r) {
          const float v = acc[r] + sRed[nf * 256 + r * 32 + lane];
          const int m = qBase + r + 8 * half;
          const int feat = h * 64 + nf * 16 + m16;
          ctx[((size_t)b * KSEL + m) * D_ + feat] = f2bf(v);
        }
      }
      __syncthreads();   // protect sS/sP before next head
    }
  }
}

// ---------------------------------------------------------------------------
// K6: out projection: out = ctx @ Wo^T + bo.  M=8192, N=1024, K=1024.
// 128x128x32 tiling, double-buffered via CDNA5 async global->LDS B128 loads:
// stage k+1 tiles stream into LDS (ASYNCcnt) while WMMAs consume stage k.
// Per thread per stage: 4 async B128s -> s_wait_asynccnt 4 retires exactly
// the current stage (async loads complete in order).
// ---------------------------------------------------------------------------
__global__ void __launch_bounds__(256) out_gemm_kernel(
    const uint16_t* __restrict__ A,    // [8192][1024] bf16 (ctx)
    const uint16_t* __restrict__ Wt,   // [1024][1024] bf16 (out_proj_w, n-major)
    const float* __restrict__ bias, float* __restrict__ out) {
  __shared__ uint16_t aL[2][128][40];
  __shared__ uint16_t bL[2][128][40];
  const uint32_t BUFB = 128 * 40 * 2;  // bytes per buffer

  const int tileM = blockIdx.x;        // 0..63
  const int tileN = blockIdx.y;        // 0..7
  const int tid  = threadIdx.x;
  const int lane = tid & 31, wave = tid >> 5;
  const int m16  = lane & 15, half = lane >> 4;
  const int mw   = wave >> 2, nw = wave & 3;

  const int rowA = tid >> 1;
  const int colA = (tid & 1) * 16;
  const uint16_t* __restrict__ arow = A  + (size_t)(tileM * 128 + rowA) * D_;
  const uint16_t* __restrict__ wrow = Wt + (size_t)(tileN * 128 + rowA) * D_;
  const uint32_t aDst = (uint32_t)(uintptr_t)&aL[0][rowA][colA];
  const uint32_t bDst = (uint32_t)(uintptr_t)&bL[0][rowA][colA];

  v8f acc[4][2];
  #pragma unroll
  for (int i = 0; i < 4; ++i)
    #pragma unroll
    for (int j = 0; j < 2; ++j) acc[i][j] = zero_v8f();

  // prologue: stage 0 in flight
  async_copy_32B(aDst, arow + colA);
  async_copy_32B(bDst, wrow + colA);

  #pragma unroll 2
  for (int ks = 0; ks < 32; ++ks) {
    const int cur = ks & 1;
    if (ks < 31) {
      const int k0n = (ks + 1) * 32;
      const uint32_t nb = (uint32_t)(cur ^ 1) * BUFB;
      async_copy_32B(aDst + nb, arow + k0n + colA);
      async_copy_32B(bDst + nb, wrow + k0n + colA);
      wait_async_le4();      // current stage's 4 loads retired (in-order)
    } else {
      wait_async_0();
    }
    __syncthreads();

    #pragma unroll
    for (int nf = 0; nf < 2; ++nf) {
      const uint16_t* bp = &bL[cur][nw * 32 + nf * 16 + m16][8 * half];
      v16bf bf = load_frag(bp, bp + 16);
      #pragma unroll
      for (int mf = 0; mf < 4; ++mf) {
        const uint16_t* apf = &aL[cur][mw * 64 + mf * 16 + m16][8 * half];
        v16bf af = load_frag(apf, apf + 16);
        acc[mf][nf] = wmma_bf16(af, bf, acc[mf][nf]);
      }
    }
    __syncthreads();   // all waves done reading `cur` before it is refilled
  }

  #pragma unroll
  for (int mf = 0; mf < 4; ++mf)
    #pragma unroll
    for (int nf = 0; nf < 2; ++nf) {
      const int nGlob = tileN * 128 + nw * 32 + nf * 16 + m16;
      const float bv = bias[nGlob];
      #pragma unroll
      for (int r = 0; r < 8; ++r) {
        const int m = tileM * 128 + mw * 64 + mf * 16 + 8 * half + r;
        out[(size_t)m * D_ + nGlob] = acc[mf][nf][r] + bv;
      }
    }
}

// ---------------------------------------------------------------------------
extern "C" void kernel_launch(void* const* d_in, const int* in_sizes, int n_in,
                              void* d_out, int out_size, void* d_ws,
                              size_t ws_size, hipStream_t stream) {
  const float* query      = (const float*)d_in[0];
  const float* key        = (const float*)d_in[1];
  const float* value      = (const float*)d_in[2];
  const float* in_proj_w  = (const float*)d_in[3];
  const float* in_proj_b  = (const float*)d_in[4];
  const float* out_proj_w = (const float*)d_in[5];
  const float* out_proj_b = (const float*)d_in[6];

  // workspace carve-up (~76 MB)
  char* ws = (char*)d_ws;
  size_t off = 0;
  auto carve = [&](size_t bytes) -> void* {
    void* p = ws + off;
    off = (off + bytes + 255) & ~(size_t)255;
    return p;
  };
  float*    norms = (float*)   carve((size_t)B_ * S_ * 4);
  int*      idx   = (int*)     carve((size_t)B_ * KSEL * 4);
  uint16_t* wIn   = (uint16_t*)carve((size_t)3 * D_ * D_ * 2);
  uint16_t* wOut  = (uint16_t*)carve((size_t)D_ * D_ * 2);
  uint16_t* qp    = (uint16_t*)carve((size_t)B_ * H_ * KSEL * DH_ * 2);
  uint16_t* kp    = (uint16_t*)carve((size_t)B_ * H_ * KSEL * DH_ * 2);
  uint16_t* vt    = (uint16_t*)carve((size_t)B_ * H_ * DH_ * KSEL * 2);
  uint16_t* ctx   = (uint16_t*)carve((size_t)B_ * KSEL * D_ * 2);

  float* out   = (float*)d_out;                       // (8,1024,1024)
  float* attnw = out + (size_t)B_ * KSEL * D_;        // (8,1024,1024)

  norms_kernel<<<(B_ * S_) / 8, 256, 0, stream>>>(query, norms);
  topk_kernel<<<B_, 1024, 0, stream>>>(norms, idx);
  convert_bf16_kernel<<<(3 * D_ * D_ + 255) / 256, 256, 0, stream>>>(
      in_proj_w, wIn, 3 * D_ * D_);
  convert_bf16_kernel<<<(D_ * D_ + 255) / 256, 256, 0, stream>>>(
      out_proj_w, wOut, D_ * D_);
  proj_gemm_kernel<<<dim3(8, 24, 8), 256, 0, stream>>>(
      query, key, value, idx, wIn, in_proj_b, qp, kp, vt);
  const size_t attnSmem = 16 * SSTR * 4 + 16 * PSTR * 2 + 4 * 256 * 4;
  attn_kernel<<<dim3(64, 8), 256, attnSmem, stream>>>(qp, kp, vt, ctx, attnw);
  out_gemm_kernel<<<dim3(64, 8), 256, 0, stream>>>(ctx, wOut, out_proj_b, out);
}
